// StridedLatentReducer_64759516889545
// MI455X (gfx1250) — matrix operand order repaired and checked by reading, not compiled, not run on hardware
//
#include <hip/hip_runtime.h>
#include <hip/hip_bf16.h>
#include <math.h>

typedef __bf16 bf16_t;
typedef __attribute__((ext_vector_type(16))) __bf16 v16bf;
typedef __attribute__((ext_vector_type(8)))  __bf16 v8bf;
typedef __attribute__((ext_vector_type(8)))  float  v8f;

#define B_      8
#define L_      4096
#define C_      512
#define LO_     2048              // L/2 conv output length
#define M_      (B_ * LO_)        // 16384 GEMM rows
#define SLOTS_  8
#define OUTROWS_ (SLOTS_ + LO_)   // 2056

// ---------------- workspace layout (bytes) ----------------
static const size_t OFF_H   = 0;                                      // h bf16 [M_, C_]
static const size_t SZ_H    = (size_t)M_ * C_ * sizeof(bf16_t);       // 16 MiB
static const size_t OFF_W   = OFF_H + SZ_H;                           // pw_w bf16 [N, K]
static const size_t SZ_W    = (size_t)C_ * C_ * sizeof(bf16_t);       // 512 KiB
static const size_t OFF_SC  = OFF_W + SZ_W;                           // scores f32 [B_, L_]
static const size_t SZ_SC   = (size_t)B_ * L_ * sizeof(float);        // 128 KiB
static const size_t OFF_IX  = OFF_SC + SZ_SC;                         // idx int [B_, 8]
static const size_t SZ_IX   = 256;
static const size_t OFF_RED = OFF_IX + SZ_IX;                         // reduced f32 [M_, C_] 32 MiB

// ---------------- helpers ----------------
static __device__ __forceinline__ float wred(float v) {
#pragma unroll
  for (int off = 16; off > 0; off >>= 1) v += __shfl_xor(v, off, 32);
  return v;
}

static __device__ __forceinline__ void load16(const float* __restrict__ p, float* x) {
  const float4* q = (const float4*)p;
#pragma unroll
  for (int i = 0; i < 4; ++i) {
    float4 t = q[i];
    x[4*i+0] = t.x; x[4*i+1] = t.y; x[4*i+2] = t.z; x[4*i+3] = t.w;
  }
}

static __device__ __forceinline__ v16bf cat8(v8bf a, v8bf b) {
  v16bf r;
#pragma unroll
  for (int i = 0; i < 8; ++i) { r[i] = a[i]; r[8+i] = b[i]; }
  return r;
}

// ---------------- K1: LN1 + depthwise conv(s=2,k=3) + GELU -> h bf16; + carrier scores ----------------
__global__ void __launch_bounds__(256)
k_ln1_conv(const float* __restrict__ latents,
           const float* __restrict__ ln1_g, const float* __restrict__ ln1_b,
           const float* __restrict__ dw_w,  const float* __restrict__ dw_b,
           const float* __restrict__ carrier_w, const float* __restrict__ carrier_b,
           bf16_t* __restrict__ h_out, float* __restrict__ scores)
{
  const int wid  = blockIdx.x * (blockDim.x >> 5) + (threadIdx.x >> 5);
  const int lane = threadIdx.x & 31;
  const int b    = wid >> 11;
  const int lo   = wid & (LO_ - 1);
  const int r    = lo * 2;
  const int c0   = lane * 16;
  const float* base = latents + ((size_t)b * L_) * C_;

  float xm[16], x0[16], xp[16];
  const bool have_m1 = (r > 0);
  load16(base + (size_t)r       * C_ + c0, x0);
  load16(base + (size_t)(r + 1) * C_ + c0, xp);
  if (have_m1) load16(base + (size_t)(r - 1) * C_ + c0, xm);
  else {
#pragma unroll
    for (int i = 0; i < 16; ++i) xm[i] = 0.f;
  }

  float s0 = 0.f, q0 = 0.f, s1 = 0.f, q1 = 0.f, sm = 0.f, qm = 0.f;
#pragma unroll
  for (int i = 0; i < 16; ++i) {
    s0 += x0[i]; q0 += x0[i] * x0[i];
    s1 += xp[i]; q1 += xp[i] * xp[i];
    sm += xm[i]; qm += xm[i] * xm[i];
  }
  s0 = wred(s0); q0 = wred(q0);
  s1 = wred(s1); q1 = wred(q1);
  sm = wred(sm); qm = wred(qm);
  const float inv = 1.0f / (float)C_;
  const float mu0 = s0 * inv, mu1 = s1 * inv, mum = sm * inv;
  const float rs0 = rsqrtf(q0 * inv - mu0 * mu0 + 1e-5f);
  const float rs1 = rsqrtf(q1 * inv - mu1 * mu1 + 1e-5f);
  const float rsm = have_m1 ? rsqrtf(qm * inv - mum * mum + 1e-5f) : 0.f;

  float g[16], bb[16], cw[16], w0[16], w1[16], w2[16], db[16];
#pragma unroll
  for (int i = 0; i < 16; ++i) {
    const int c = c0 + i;
    g[i]  = ln1_g[c];  bb[i] = ln1_b[c];  cw[i] = carrier_w[c];
    w0[i] = dw_w[3*c]; w1[i] = dw_w[3*c + 1]; w2[i] = dw_w[3*c + 2];
    db[i] = dw_b[c];
  }

  float sc0 = 0.f, sc1 = 0.f;
  v8bf hlo{}, hhi{};
#pragma unroll
  for (int i = 0; i < 16; ++i) {
    const float n0 = (x0[i] - mu0) * rs0 * g[i] + bb[i];
    const float np = (xp[i] - mu1) * rs1 * g[i] + bb[i];
    const float nm = have_m1 ? ((xm[i] - mum) * rsm * g[i] + bb[i]) : 0.f;
    sc0 += n0 * cw[i];
    sc1 += np * cw[i];
    float h = nm * w0[i] + n0 * w1[i] + np * w2[i] + db[i];
    h = 0.5f * h * (1.0f + erff(h * 0.70710678118654752f));   // exact GELU
    if (i < 8) hlo[i] = (bf16_t)h; else hhi[i - 8] = (bf16_t)h;
  }
  sc0 = wred(sc0); sc1 = wred(sc1);
  if (lane == 0) {
    const float cb = carrier_b[0];
    scores[(size_t)b * L_ + r]     = sc0 + cb;  // even rows: written once
    scores[(size_t)b * L_ + r + 1] = sc1 + cb;  // odd rows: written once
  }
  bf16_t* hp = h_out + (size_t)wid * C_ + c0;
  *(v8bf*)hp       = hlo;
  *(v8bf*)(hp + 8) = hhi;
}

// ---------------- K2: pw_w f32 -> bf16 [N,K] ----------------
__global__ void __launch_bounds__(256)
k_wconv(const float* __restrict__ pw_w, bf16_t* __restrict__ wbf) {
  const int i = blockIdx.x * blockDim.x + threadIdx.x;   // grid covers exactly C_*C_
  wbf[i] = (bf16_t)pw_w[i];
}

// ---------------- K3: per-batch top-8 (one wave per batch) ----------------
__global__ void k_topk(const float* __restrict__ scores, int* __restrict__ idx_out) {
  const int b = blockIdx.x;
  const int lane = threadIdx.x;     // blockDim == 32
  const float* s = scores + (size_t)b * L_;
  int chosen[SLOTS_];
#pragma unroll
  for (int k = 0; k < SLOTS_; ++k) chosen[k] = -1;

  for (int k = 0; k < SLOTS_; ++k) {
    float bestV = -INFINITY; int bestI = 0x7fffffff;
    for (int j = lane; j < L_; j += 32) {
      bool used = false;
#pragma unroll
      for (int t = 0; t < SLOTS_; ++t) used |= (chosen[t] == j);
      const float v = used ? -INFINITY : s[j];
      if (v > bestV || (v == bestV && j < bestI)) { bestV = v; bestI = j; }
    }
#pragma unroll
    for (int off = 16; off > 0; off >>= 1) {
      const float ov = __shfl_xor(bestV, off, 32);
      const int   oi = __shfl_xor(bestI, off, 32);
      if (ov > bestV || (ov == bestV && oi < bestI)) { bestV = ov; bestI = oi; }
    }
    chosen[k] = bestI;              // all lanes agree after butterfly
  }
  // sort ascending (reference sorts selected indices)
#pragma unroll
  for (int i = 0; i < SLOTS_; ++i)
#pragma unroll
    for (int j = i + 1; j < SLOTS_; ++j)
      if (chosen[j] < chosen[i]) { int t = chosen[i]; chosen[i] = chosen[j]; chosen[j] = t; }
  if (lane == 0) {
#pragma unroll
    for (int k = 0; k < SLOTS_; ++k) idx_out[b * SLOTS_ + k] = chosen[k];
  }
}

// ---------------- K4: reduced = h @ pw_w^T + pw_b via v_wmma_f32_16x16x32_bf16 ----------------
// One wave computes a 64(M) x 64(N) tile: 16 accumulators.
// Per K-step: 4 B-fragments + 4 A-fragments (16 b128 loads) feed 16 WMMAs (1 load/WMMA);
// A-fragment reused 4x in registers, B-fragments reused 4x.
__global__ void __launch_bounds__(256)
k_gemm(const bf16_t* __restrict__ hmat, const bf16_t* __restrict__ wmat,
       const float* __restrict__ pw_b, float* __restrict__ reduced)
{
  const int lane   = threadIdx.x & 31;
  const int wave   = threadIdx.x >> 5;
  const int waveId = blockIdx.x * (blockDim.x >> 5) + wave;
  const int ng   = waveId & 7;         // 8 N-groups of 64 cols
  const int mg   = waveId >> 3;        // 256 M-groups of 64 rows
  const int m0   = mg * 64;
  const int n0   = ng * 64;
  const int half = lane >> 4;
  const int lan  = lane & 15;

  v8f acc[4][4] = {};                   // [m-tile][n-tile]
  const bf16_t* wbase[4];
#pragma unroll
  for (int j = 0; j < 4; ++j) wbase[j] = wmat + (size_t)(n0 + j * 16 + lan) * C_;
  const int arow0 = m0 + lan;

  for (int kk = 0; kk < 16; ++kk) {
    // B fragments (32x16 bf16): lane half 0 -> K 0..7 & 16..23, half 1 -> K 8..15 & 24..31
    const int kb_b = kk * 32 + half * 8;
    v16bf bfrag[4];
#pragma unroll
    for (int j = 0; j < 4; ++j) {
      const v8bf b0 = *(const v8bf*)(wbase[j] + kb_b);
      const v8bf b1 = *(const v8bf*)(wbase[j] + kb_b + 16);
      bfrag[j] = cat8(b0, b1);
    }
    // A fragments (16x32 bf16): lane half 0 -> K 0..7 & 16..23, half 1 -> K 8..15 & 24..31
    const int kb_a = kk * 32 + half * 8;
    if (kk + 1 < 16)
      __builtin_prefetch(hmat + (size_t)arow0 * C_ + (kk + 1) * 32 + half * 8, 0, 1);
#pragma unroll
    for (int t = 0; t < 4; ++t) {
      const bf16_t* ap = hmat + (size_t)(arow0 + t * 16) * C_ + kb_a;
      const v8bf a0 = *(const v8bf*)ap;
      const v8bf a1 = *(const v8bf*)(ap + 16);
      const v16bf af = cat8(a0, a1);
#pragma unroll
      for (int j = 0; j < 4; ++j)
        acc[t][j] = __builtin_amdgcn_wmma_f32_16x16x32_bf16(
            false, af, false, bfrag[j], (short)0, acc[t][j], false, false);
    }
  }

#pragma unroll
  for (int t = 0; t < 4; ++t) {
    const int mb = m0 + t * 16 + half * 8;  // C/D: VGPR r -> (M = half*8 + r, N = lan)
#pragma unroll
    for (int j = 0; j < 4; ++j) {
      const int n = n0 + j * 16 + lan;
      const float bias = pw_b[n];
#pragma unroll
      for (int rr = 0; rr < 8; ++rr)
        reduced[(size_t)(mb + rr) * C_ + n] = acc[t][j][rr] + bias;
    }
  }
}

// ---------------- K5: gate + merge + LN2 (one wave per output position) ----------------
__global__ void __launch_bounds__(256)
k_merge(const float* __restrict__ latents, const float* __restrict__ reduced,
        const float* __restrict__ gate_w, const float* __restrict__ gate_b,
        const float* __restrict__ ln2_g, const float* __restrict__ ln2_b,
        float* __restrict__ out, float* __restrict__ gate_out)
{
  const int wid  = blockIdx.x * (blockDim.x >> 5) + (threadIdx.x >> 5);
  const int lane = threadIdx.x & 31;
  const int b  = wid >> 11;
  const int lo = wid & (LO_ - 1);
  const int c0 = lane * 16;

  float rd[16], la[16], lb2[16];
  load16(reduced + (size_t)wid * C_ + c0, rd);
  const float* base = latents + ((size_t)b * L_ + 2 * lo) * C_;
  load16(base,        la);
  load16(base + C_,   lb2);

  float pooled[16];
  float dot = 0.f;
#pragma unroll
  for (int i = 0; i < 16; ++i) {
    pooled[i] = 0.5f * (la[i] + lb2[i]);
    dot += rd[i] * gate_w[c0 + i] + pooled[i] * gate_w[C_ + c0 + i];
  }
  dot = wred(dot);
  const float logit = dot + gate_b[0];
  const float gte = 0.5f + 0.5f / (1.0f + expf(-logit));

  float merged[16];
  float s = 0.f, q = 0.f;
#pragma unroll
  for (int i = 0; i < 16; ++i) {
    merged[i] = pooled[i] + gte * (rd[i] - pooled[i]);
    s += merged[i]; q += merged[i] * merged[i];
  }
  s = wred(s); q = wred(q);
  const float inv = 1.0f / (float)C_;
  const float mu = s * inv;
  const float rs = rsqrtf(q * inv - mu * mu + 1e-5f);

  float* op = out + ((size_t)b * OUTROWS_ + SLOTS_ + lo) * C_ + c0;
#pragma unroll
  for (int i = 0; i < 16; ++i)
    op[i] = (merged[i] - mu) * rs * ln2_g[c0 + i] + ln2_b[c0 + i];

  if (lane == 0) gate_out[(size_t)b * LO_ + lo] = gte;
}

// ---------------- K6: carriers gather + LN2 (one wave per slot) ----------------
__global__ void __launch_bounds__(256)
k_carriers(const float* __restrict__ latents, const int* __restrict__ idx,
           const float* __restrict__ ln2_g, const float* __restrict__ ln2_b,
           float* __restrict__ out)
{
  const int b    = blockIdx.x;
  const int slot = threadIdx.x >> 5;
  const int lane = threadIdx.x & 31;
  const int c0   = lane * 16;
  const int row  = idx[b * SLOTS_ + slot];

  float x[16];
  load16(latents + ((size_t)b * L_ + row) * C_ + c0, x);
  float s = 0.f, q = 0.f;
#pragma unroll
  for (int i = 0; i < 16; ++i) { s += x[i]; q += x[i] * x[i]; }
  s = wred(s); q = wred(q);
  const float inv = 1.0f / (float)C_;
  const float mu = s * inv;
  const float rs = rsqrtf(q * inv - mu * mu + 1e-5f);

  float* op = out + ((size_t)b * OUTROWS_ + slot) * C_ + c0;
#pragma unroll
  for (int i = 0; i < 16; ++i)
    op[i] = (x[i] - mu) * rs * ln2_g[c0 + i] + ln2_b[c0 + i];
}

// ---------------- launch ----------------
extern "C" void kernel_launch(void* const* d_in, const int* in_sizes, int n_in,
                              void* d_out, int out_size, void* d_ws, size_t ws_size,
                              hipStream_t stream) {
  (void)in_sizes; (void)n_in; (void)out_size; (void)ws_size;
  const float* latents   = (const float*)d_in[0];
  const float* ln1_g     = (const float*)d_in[1];
  const float* ln1_b     = (const float*)d_in[2];
  const float* dw_w      = (const float*)d_in[3];
  const float* dw_b      = (const float*)d_in[4];
  const float* pw_w      = (const float*)d_in[5];
  const float* pw_b      = (const float*)d_in[6];
  const float* gate_w    = (const float*)d_in[7];
  const float* gate_b    = (const float*)d_in[8];
  const float* carrier_w = (const float*)d_in[9];
  const float* carrier_b = (const float*)d_in[10];
  const float* ln2_g     = (const float*)d_in[11];
  const float* ln2_b     = (const float*)d_in[12];
  // d_in[13] = carrier_slots (=8, hardcoded)

  char* ws = (char*)d_ws;
  bf16_t* h_bf    = (bf16_t*)(ws + OFF_H);
  bf16_t* w_bf    = (bf16_t*)(ws + OFF_W);
  float*  scores  = (float*) (ws + OFF_SC);
  int*    idxp    = (int*)   (ws + OFF_IX);
  float*  reduced = (float*) (ws + OFF_RED);

  float* out      = (float*)d_out;
  float* gate_out = out + (size_t)B_ * OUTROWS_ * C_;

  k_ln1_conv<<<M_ / 8, 256, 0, stream>>>(latents, ln1_g, ln1_b, dw_w, dw_b,
                                         carrier_w, carrier_b, h_bf, scores);
  k_wconv<<<(C_ * C_) / 256, 256, 0, stream>>>(pw_w, w_bf);
  k_topk<<<B_, 32, 0, stream>>>(scores, idxp);
  // 256 M-groups x 8 N-groups = 2048 waves, 8 waves/block -> 256 blocks
  k_gemm<<<256, 256, 0, stream>>>(h_bf, w_bf, pw_b, reduced);
  k_merge<<<M_ / 8, 256, 0, stream>>>(latents, reduced, gate_w, gate_b,
                                      ln2_g, ln2_b, out, gate_out);
  k_carriers<<<B_, 256, 0, stream>>>(latents, idxp, ln2_g, ln2_b, out);
}
